// AmazonGCN_85899345920152
// MI455X (gfx1250) — matrix operand order repaired and checked by reading, not compiled, not run on hardware
//
#include <hip/hip_runtime.h>
#include <hip/hip_bf16.h>
#include <stdint.h>

// ---------------------------------------------------------------------------
// GCN 2-layer forward for gfx1250 (MI455X).
//   W1t = transpose(W1) packed bf16 (one tiny prep kernel, reused by all blocks)
//   h1  = x @ W1        (WMMA bf16/f32 accum; W1t -> LDS via TDM tensor_load_to_lds)
//   g1  = scatter-add(norm * h1[src]) + selfloop + b1
//   z   = relu(g1) @ W2 (padded to 16 cols, WMMA)
//   g2  = scatter-add(norm * z[src])  + selfloop + b2
//   out = log_softmax(g2)
// ---------------------------------------------------------------------------

typedef __bf16 v16bf __attribute__((ext_vector_type(16)));
typedef __bf16 v8bf  __attribute__((ext_vector_type(8)));
typedef float  v8f   __attribute__((ext_vector_type(8)));
typedef unsigned int u32x4 __attribute__((ext_vector_type(4)));
typedef unsigned int u32x8 __attribute__((ext_vector_type(8)));

#define F_IN 512
#define HID  64
#define NCLS 10
#define NPAD 16

__device__ __forceinline__ v8f wmma_bf16(v16bf a, v16bf b, v8f c) {
    // (neg_a, A, neg_b, B, c_mod, C, reuse_a, reuse_b)
    return __builtin_amdgcn_wmma_f32_16x16x32_bf16(false, a, false, b, (short)0, c,
                                                   false, false);
}

// --------------------------- degree / norm ---------------------------------
__global__ void deg_init_k(float* deg, int N) {
    int i = blockIdx.x * blockDim.x + threadIdx.x;
    if (i < N) deg[i] = 1.0f;   // self-loop contributes 1 to each target degree
}

__global__ void deg_accum_k(const long long* __restrict__ dst, float* deg, long long E) {
    long long e = (long long)blockIdx.x * blockDim.x + threadIdx.x;
    if (e < E) atomicAdd(&deg[(int)dst[e]], 1.0f);
}

__global__ void deg_rsqrt_k(float* deg, int N) {
    int i = blockIdx.x * blockDim.x + threadIdx.x;
    if (i < N) deg[i] = rsqrtf(deg[i]);   // deg >= 1 always (self-loops)
}

// ---------------- W1 transpose + bf16 pack: W1t[col*512 + k] ---------------
__global__ void w1_pack_k(const float* __restrict__ W1, __bf16* __restrict__ W1t) {
    int idx = blockIdx.x * blockDim.x + threadIdx.x;   // grid covers 64*512
    if (idx < HID * F_IN) {
        int col = idx >> 9;       // /512
        int k   = idx & 511;
        W1t[idx] = (__bf16)W1[k * HID + col];
    }
}

// --------------------------- layer-1 GEMM (WMMA + TDM) ---------------------
// Block = 128 threads (4 waves). Each wave computes a 16x64 tile of h1.
// W1t (bf16, already transposed) is DMA'd into LDS by the Tensor Data Mover.
__global__ __launch_bounds__(128) void gemm1_k(const float* __restrict__ x,
                                               const __bf16* __restrict__ W1t,
                                               float* __restrict__ h1, int N) {
    __shared__ __bf16 Wlds[HID * F_IN];   // 64 KB

    // LDS byte offset of Wlds: generic LDS address = {SHARED_BASE, offset[31:0]}.
    // Passing this into the asm also makes Wlds "escape", so the compiler cannot
    // fold the later ds_loads to undef (the TDM writes LDS behind its back).
    unsigned ldsAddr = (unsigned)(unsigned long long)(uintptr_t)&Wlds[0];

    if ((threadIdx.x >> 5) == 0) {        // one wave issues the TDM descriptor
        unsigned long long ga = (unsigned long long)(uintptr_t)W1t;
        const unsigned elems = HID * F_IN;          // 32768 bf16 = 64 KB
        u32x4 g0;
        g0.x = 1u;                                  // count=1, user mode, no gather
        g0.y = ldsAddr;                             // lds_addr (bytes)
        g0.z = (unsigned)(ga & 0xffffffffu);        // global_addr[31:0]
        g0.w = (unsigned)((ga >> 32) & 0x01ffffffu) // global_addr[56:32]
               | 0x80000000u;                       // type = 2 ("image")
        u32x8 g1v;
        g1v[0] = (1u << 16);                        // data_size=1 -> 2 bytes; wg_mask=0
        g1v[1] = (elems & 0xffffu) << 16;           // tensor_dim0[15:0]
        g1v[2] = (elems >> 16) | (1u << 16);        // tensor_dim0[31:16] | tensor_dim1=1
        g1v[3] = (elems & 0xffffu) << 16;           // tile_dim0 = 32768 (0x8000)
        g1v[4] = 0u;                                // tile_dim1=0, tile_dim2=0
        g1v[5] = elems;                             // tensor_dim0_stride[31:0]
        g1v[6] = 0u;
        g1v[7] = 0u;
        asm volatile("tensor_load_to_lds %0, %1\n\t"
                     "s_wait_tensorcnt 0x0"
                     :
                     : "s"(g0), "s"(g1v)
                     : "memory");
    }
    __syncthreads();

    const int lane    = threadIdx.x & 31;
    const int wave    = threadIdx.x >> 5;
    const int rowBase = blockIdx.x * 64 + wave * 16;
    const int mrow    = rowBase + (lane & 15);
    const int arow    = (mrow < N) ? mrow : (N - 1);   // clamp: EXEC stays all-ones
    const int kHalfA  = (lane >> 4) * 16;              // A: K=0..15 | 16..31
    const int kSelB   = (lane >> 4) * 16;              // B: K=0..15 | 16..31
    const int colB    = lane & 15;

    const float* xrow = x + (size_t)arow * F_IN;

    v8f acc[4] = {};
    for (int k0 = 0; k0 < F_IN; k0 += 32) {
        __builtin_prefetch(xrow + k0 + 128, 0, 0);     // global_prefetch_b8

        v16bf a;
#pragma unroll
        for (int j = 0; j < 8; ++j) {
            a[j]     = (__bf16)xrow[k0 + kHalfA + j];
            a[8 + j] = (__bf16)xrow[k0 + 8 + kHalfA + j];
        }
#pragma unroll
        for (int n = 0; n < 4; ++n) {
            const __bf16* bp = &Wlds[(size_t)(n * 16 + colB) * F_IN + k0 + kSelB];
            const v8bf* bp8 = reinterpret_cast<const v8bf*>(bp);
            v8bf blo = bp8[0], bhi = bp8[1];
            v16bf b;
#pragma unroll
            for (int j = 0; j < 8; ++j) { b[j] = blo[j]; b[8 + j] = bhi[j]; }
            acc[n] = wmma_bf16(a, b, acc[n]);
        }
    }

    // C/D layout: VGPR r -> M = r (lanes 0-15) or 8+r (lanes 16-31); N = lane&15
    if (rowBase + 16 <= N) {               // uniform full-tile fast path
#pragma unroll
        for (int r = 0; r < 8; ++r) {
            int row = rowBase + (lane >> 4) * 8 + r;
            float* hp = h1 + (size_t)row * HID + colB;
#pragma unroll
            for (int n = 0; n < 4; ++n) hp[n * 16] = acc[n][r];
        }
    } else {
#pragma unroll
        for (int r = 0; r < 8; ++r) {
            int row = rowBase + (lane >> 4) * 8 + r;
            if (row < N) {
                float* hp = h1 + (size_t)row * HID + colB;
#pragma unroll
                for (int n = 0; n < 4; ++n) hp[n * 16] = acc[n][r];
            }
        }
    }
}

// ------------------- layer-1 aggregation (self-loop + bias, then edges) ----
__global__ void agg1_init_k(const float* __restrict__ h1,
                            const float* __restrict__ dinv,
                            const float* __restrict__ b1,
                            float* __restrict__ g1, int N) {
    long long idx = (long long)blockIdx.x * blockDim.x + threadIdx.x;
    if (idx < (long long)N * HID) {
        int i = (int)(idx >> 6), f = (int)(idx & 63);
        float di = dinv[i];
        g1[idx] = h1[idx] * di * di + b1[f];
    }
}

// warp per edge, 2 features per lane (coalesced 256B gather)
__global__ __launch_bounds__(256) void agg1_edges_k(const float* __restrict__ h1,
                                                    const long long* __restrict__ src,
                                                    const long long* __restrict__ dst,
                                                    const float* __restrict__ dinv,
                                                    float* __restrict__ g1, long long E) {
    long long e = (long long)blockIdx.x * 8 + (threadIdx.x >> 5);
    if (e >= E) return;
    int s = (int)src[e], d = (int)dst[e];
    float nrm = dinv[s] * dinv[d];
    int f = (threadIdx.x & 31) * 2;
    const float2 hv = *reinterpret_cast<const float2*>(&h1[(size_t)s * HID + f]);
    atomicAdd(&g1[(size_t)d * HID + f],     hv.x * nrm);
    atomicAdd(&g1[(size_t)d * HID + f + 1], hv.y * nrm);
}

// --------------------------- layer-2 GEMM (WMMA, relu fused) ---------------
// Block = 256 threads (8 waves). Each wave computes a 16x16 tile of z.
__global__ __launch_bounds__(256) void gemm2_k(const float* __restrict__ g1,
                                               const float* __restrict__ W2,
                                               float* __restrict__ z, int N) {
    __shared__ __bf16 Wlds2[NPAD * HID];   // [col*64 + k], cols 10..15 zero
    for (int idx = threadIdx.x; idx < NPAD * HID; idx += 256) {
        int col = idx >> 6, k = idx & 63;
        Wlds2[idx] = (col < NCLS) ? (__bf16)W2[k * NCLS + col] : (__bf16)0.0f;
    }
    __syncthreads();

    const int lane    = threadIdx.x & 31;
    const int wave    = threadIdx.x >> 5;
    const int rowBase = blockIdx.x * 128 + wave * 16;
    const int mrow    = rowBase + (lane & 15);
    const int arow    = (mrow < N) ? mrow : (N - 1);
    const int kHalfA  = (lane >> 4) * 16;
    const int kSelB   = (lane >> 4) * 16;
    const int colB    = lane & 15;

    const float* grow = g1 + (size_t)arow * HID;

    v8f acc = {};
#pragma unroll
    for (int k0 = 0; k0 < HID; k0 += 32) {
        v16bf a;
#pragma unroll
        for (int j = 0; j < 8; ++j) {
            a[j]     = (__bf16)fmaxf(grow[k0 + kHalfA + j], 0.0f);       // relu
            a[8 + j] = (__bf16)fmaxf(grow[k0 + 8 + kHalfA + j], 0.0f);  // relu
        }
        const v8bf* bp8 = reinterpret_cast<const v8bf*>(&Wlds2[colB * HID + k0 + kSelB]);
        v8bf blo = bp8[0], bhi = bp8[1];
        v16bf b;
#pragma unroll
        for (int j = 0; j < 8; ++j) { b[j] = blo[j]; b[8 + j] = bhi[j]; }
        acc = wmma_bf16(a, b, acc);
    }

    if (rowBase + 16 <= N) {
#pragma unroll
        for (int r = 0; r < 8; ++r) {
            int row = rowBase + (lane >> 4) * 8 + r;
            z[(size_t)row * NPAD + colB] = acc[r];
        }
    } else {
#pragma unroll
        for (int r = 0; r < 8; ++r) {
            int row = rowBase + (lane >> 4) * 8 + r;
            if (row < N) z[(size_t)row * NPAD + colB] = acc[r];
        }
    }
}

// ------------------- layer-2 aggregation + log_softmax ---------------------
__global__ void agg2_init_k(const float* __restrict__ z,
                            const float* __restrict__ dinv,
                            const float* __restrict__ b2,
                            float* __restrict__ g2, int N) {
    long long idx = (long long)blockIdx.x * blockDim.x + threadIdx.x;
    if (idx < (long long)N * NCLS) {
        int i = (int)(idx / NCLS), c = (int)(idx % NCLS);
        float di = dinv[i];
        g2[idx] = z[(size_t)i * NPAD + c] * di * di + b2[c];
    }
}

__global__ void agg2_edges_k(const float* __restrict__ z,
                             const long long* __restrict__ src,
                             const long long* __restrict__ dst,
                             const float* __restrict__ dinv,
                             float* __restrict__ g2, long long E) {
    long long e = (long long)blockIdx.x * blockDim.x + threadIdx.x;
    if (e >= E) return;
    int s = (int)src[e], d = (int)dst[e];
    float nrm = dinv[s] * dinv[d];
    const float* zs = z + (size_t)s * NPAD;
    float* gd = g2 + (size_t)d * NCLS;
#pragma unroll
    for (int c = 0; c < NCLS; ++c) atomicAdd(&gd[c], zs[c] * nrm);
}

__global__ void logsoftmax_k(const float* __restrict__ g2, float* __restrict__ out, int N) {
    int i = blockIdx.x * blockDim.x + threadIdx.x;
    if (i >= N) return;
    float v[NCLS];
    float m = -3.402823466e+38f;
#pragma unroll
    for (int c = 0; c < NCLS; ++c) { v[c] = g2[(size_t)i * NCLS + c]; m = fmaxf(m, v[c]); }
    float s = 0.0f;
#pragma unroll
    for (int c = 0; c < NCLS; ++c) s += __expf(v[c] - m);
    float l = __logf(s);
#pragma unroll
    for (int c = 0; c < NCLS; ++c) out[(size_t)i * NCLS + c] = v[c] - m - l;
}

// ---------------------------------------------------------------------------
extern "C" void kernel_launch(void* const* d_in, const int* in_sizes, int n_in,
                              void* d_out, int out_size, void* d_ws, size_t ws_size,
                              hipStream_t stream) {
    const float*     x  = (const float*)d_in[0];
    const long long* ei = (const long long*)d_in[1];   // int64 edge_index [2,E]
    const float*     W1 = (const float*)d_in[2];
    const float*     b1 = (const float*)d_in[3];
    const float*     W2 = (const float*)d_in[4];
    const float*     b2 = (const float*)d_in[5];

    const long long E = (long long)in_sizes[1] / 2;
    const int       N = in_sizes[0] / F_IN;
    const long long* src = ei;
    const long long* dst = ei + E;

    // workspace layout: W1t[64KB as bf16] | dinv[N] | h1[N*64] | g1[N*64] | z[N*16] | g2[N*10]
    __bf16* W1t = (__bf16*)d_ws;                       // 32768 bf16 = 64 KB
    float* dinv = (float*)((char*)d_ws + (size_t)HID * F_IN * sizeof(__bf16));
    float* h1   = dinv + N;
    float* g1   = h1 + (size_t)N * HID;
    float* z    = g1 + (size_t)N * HID;
    float* g2   = z  + (size_t)N * NPAD;
    float* out  = (float*)d_out;

    // 1) symmetric normalization coefficients + weight pack (independent)
    deg_init_k <<<(N + 255) / 256, 256, 0, stream>>>(dinv, N);
    deg_accum_k<<<(unsigned)((E + 255) / 256), 256, 0, stream>>>(dst, dinv, E);
    deg_rsqrt_k<<<(N + 255) / 256, 256, 0, stream>>>(dinv, N);
    w1_pack_k  <<<(HID * F_IN + 255) / 256, 256, 0, stream>>>(W1, W1t);

    // 2) layer 1: GEMM (WMMA, TDM-staged weights) then aggregate
    gemm1_k<<<(N + 63) / 64, 128, 0, stream>>>(x, W1t, h1, N);
    agg1_init_k<<<(unsigned)(((long long)N * HID + 255) / 256), 256, 0, stream>>>(h1, dinv, b1, g1, N);
    agg1_edges_k<<<(unsigned)((E + 7) / 8), 256, 0, stream>>>(h1, src, dst, dinv, g1, E);

    // 3) layer 2: relu fused into GEMM load, then aggregate
    gemm2_k<<<(N + 127) / 128, 256, 0, stream>>>(g1, W2, z, N);
    agg2_init_k<<<(unsigned)(((long long)N * NCLS + 255) / 256), 256, 0, stream>>>(z, dinv, b2, g2, N);
    agg2_edges_k<<<(unsigned)((E + 255) / 256), 256, 0, stream>>>(z, src, dst, dinv, g2, E);

    // 4) log-softmax
    logsoftmax_k<<<(N + 255) / 256, 256, 0, stream>>>(g2, out, N);
}
